// NonStructured_Encoder_14714557956235
// MI455X (gfx1250) — matrix operand order repaired and checked by reading, not compiled, not run on hardware
//
#include <hip/hip_runtime.h>
#include <hip/hip_bf16.h>

// ---------------------------------------------------------------------------
// Problem constants (from reference)
// ---------------------------------------------------------------------------
#define DD 32
#define TT_ 32
#define LL 64
#define EE 300
#define HH 256          // per-direction hidden
#define G3 768          // 3*H gate width
#define BATCH1 (DD*TT_) // 1024 word-level sequences
#define NT_TOT 48       // 768/16 N tiles
#define KT_H 8          // 256/32 K tiles for hidden proj

typedef __attribute__((ext_vector_type(16))) __bf16 v16bf;
typedef __attribute__((ext_vector_type(8)))  __bf16 v8bf;
typedef __attribute__((ext_vector_type(8)))  float  v8f;

// packed-weight offsets in d_ws (bf16 elements)
#define SZ_IH1 (NT_TOT*10*512)   // K=300 -> pad 320 -> 10 k-tiles
#define SZ_HH  (NT_TOT*8*512)    // K=256 -> 8 k-tiles
#define SZ_IH2 (NT_TOT*16*512)   // K=512 -> 16 k-tiles
#define OFF_IH_F1 0
#define OFF_HH_F1 (OFF_IH_F1 + SZ_IH1)
#define OFF_IH_B1 (OFF_HH_F1 + SZ_HH)
#define OFF_HH_B1 (OFF_IH_B1 + SZ_IH1)
#define OFF_IH_F2 (OFF_HH_B1 + SZ_HH)
#define OFF_HH_F2 (OFF_IH_F2 + SZ_IH2)
#define OFF_IH_B2 (OFF_HH_F2 + SZ_HH)
#define OFF_HH_B2 (OFF_IH_B2 + SZ_IH2)

__device__ __forceinline__ v8f wmma_bf16(v16bf a, v16bf b, v8f c) {
  return __builtin_amdgcn_wmma_f32_16x16x32_bf16(false, a, false, b,
                                                 (short)0, c, false, false);
}

// A fragment (16x32 bf16, M x K) from an LDS row-major [16][stride] buffer.
//   lanes 0-15 : row M = lane,    K = kt*32 + {0..15}
//   lanes 16-31: row M = lane-16, K = kt*32 + {16..31}
__device__ __forceinline__ v16bf load_a_frag(const __bf16* base, int stride,
                                             int kt, int lane) {
  const __bf16* p = base + (lane & 15) * stride + kt * 32 + ((lane >> 4) << 4);
  v8bf lo = *(const v8bf*)(p);
  v8bf hi = *(const v8bf*)(p + 8);
  return __builtin_shufflevector(lo, hi, 0,1,2,3,4,5,6,7,8,9,10,11,12,13,14,15);
}

// Store a 16x16 f32 C/D tile to an LDS [16][ldc] buffer.
// VGPR j: M = j (lanes 0-15) / j+8 (lanes 16-31), N = lane&15.
__device__ __forceinline__ void store_tile(float* dst, int ldc, int ntile,
                                           int lane, v8f acc) {
  const int n  = ntile * 16 + (lane & 15);
  const int mb = (lane >> 4) << 3;
#pragma unroll
  for (int jj = 0; jj < 8; ++jj) dst[(mb + jj) * ldc + n] = acc[jj];
}

__device__ __forceinline__ float fast_sigmoid(float x) {
  return 1.0f / (1.0f + __expf(-x));
}
__device__ __forceinline__ float fast_tanh(float x) {
  float e = __expf(2.0f * x);
  return (e - 1.0f) / (e + 1.0f);
}

// ---------------------------------------------------------------------------
// Pack W[768 x Ksrc] (row-major fp32) into per-lane WMMA B-fragment layout:
// dst[(nt*KT + kt)*512 + lane*16 + e] = bf16(W[nt*16 + (lane&15)]
//                                            [kt*32 + (lane>>4)*16 + e])
// ---------------------------------------------------------------------------
__global__ __launch_bounds__(256) void pack_weight(const float* __restrict__ W,
                                                   __bf16* __restrict__ dst,
                                                   int Ksrc, int KT) {
  int total = NT_TOT * KT * 512;
  for (int i = blockIdx.x * blockDim.x + threadIdx.x; i < total;
       i += gridDim.x * blockDim.x) {
    int e    = i & 15;
    int lane = (i >> 4) & 31;
    int tile = i >> 9;
    int kt   = tile % KT;
    int nt   = tile / KT;
    int n = nt * 16 + (lane & 15);
    int k = kt * 32 + ((lane >> 4) << 4) + e;
    float v = (k < Ksrc) ? W[(size_t)n * Ksrc + k] : 0.0f;
    dst[i] = (__bf16)v;
  }
}

// ---------------------------------------------------------------------------
// One bi-GRU direction for a tile of 16 sequences.
//   grid.x = batch/16, grid.y = dir (0 fwd, 1 bwd), block = 256 (8 waves)
// Each wave owns 4 r/z N-tiles (x-proj + h-proj fused) and 2 n-gate N-tiles
// (xn and hn kept separate for n = tanh(xn + r*hn)).
// ---------------------------------------------------------------------------
template<int KIN, int KPAD, int STEPS, bool GATHER, bool WRITE_SEQ>
__global__ __launch_bounds__(256) void gru_level(
    const int*   __restrict__ tokens,
    const float* __restrict__ src,      // embed table (GATHER) or [B,STEPS,KIN]
    const __bf16* __restrict__ wih_f, const __bf16* __restrict__ wih_b,
    const __bf16* __restrict__ whh_f, const __bf16* __restrict__ whh_b,
    const float* __restrict__ bih_f, const float* __restrict__ bih_b,
    const float* __restrict__ bhh_f, const float* __restrict__ bhh_b,
    float* __restrict__ out_h,          // [B, 512], col dir*256   (!WRITE_SEQ)
    float* __restrict__ out_seq)        // [B, STEPS, 512]          (WRITE_SEQ)
{
  constexpr int KT_X = KPAD / 32;

  __shared__ float  gsum[16 * G3];      // xr+hr | xz+hz | xn     (48 KB)
  __shared__ float  hnb [16 * HH];      // hn                     (16 KB)
  __shared__ float  hF  [16 * HH];      // fp32 hidden state      (16 KB)
  __shared__ __bf16 xA  [16 * KPAD];    // bf16 input rows
  __shared__ __bf16 hA  [16 * HH];      // bf16 hidden rows

  const int tid  = threadIdx.x;
  const int wave = tid >> 5;
  const int lane = tid & 31;
  const int dir  = blockIdx.y;
  const int s0   = blockIdx.x * 16;

  const __bf16* wih = dir ? wih_b : wih_f;
  const __bf16* whh = dir ? whh_b : whh_f;
  const float*  bih = dir ? bih_b : bih_f;
  const float*  bhh = dir ? bhh_b : bhh_f;

  // per-thread gate column and hoisted biases
  const int c = tid;                    // 0..255
  const float brb = bih[c]        + bhh[c];
  const float bzb = bih[HH  + c]  + bhh[HH  + c];
  const float bxn = bih[2*HH + c];
  const float bhn = bhh[2*HH + c];

  for (int i = tid; i < 16 * HH; i += 256) {
    hF[i] = 0.0f;
    hA[i] = (__bf16)0.0f;
  }

  // tile ownership (wave-uniform)
  const int nt_rz = wave * 4;           // 4 r/z tiles: nt_rz .. nt_rz+3
  const int nt_n  = 32 + wave * 2;      // 2 n tiles
  const __bf16* bx_rz = wih + (size_t)(nt_rz * KT_X) * 512 + lane * 16;
  const __bf16* bh_rz = whh + (size_t)(nt_rz * KT_H) * 512 + lane * 16;
  const __bf16* bx_n  = wih + (size_t)(nt_n  * KT_X) * 512 + lane * 16;
  const __bf16* bh_n  = whh + (size_t)(nt_n  * KT_H) * 512 + lane * 16;

  // input staging roles: 16 rows x 16 threads per row
  const int fr = tid >> 4;              // row 0..15
  const int fc = tid & 15;              // col group

  __syncthreads();

  for (int t = 0; t < STEPS; ++t) {
    const int tt = dir ? (STEPS - 1 - t) : t;

    // ---- stage input rows for this step into LDS (bf16) ----
    if (GATHER) {
      const int tok = tokens[(size_t)(s0 + fr) * STEPS + tt];
      const float* erow = src + (size_t)tok * KIN;
      for (int k = fc; k < KPAD; k += 16)
        xA[fr * KPAD + k] = (k < KIN) ? (__bf16)erow[k] : (__bf16)0.0f;
    } else {
      const float* irow = src + ((size_t)(s0 + fr) * STEPS + tt) * KIN;
      for (int k = fc; k < KPAD; k += 16)
        xA[fr * KPAD + k] = (__bf16)irow[k];
    }
    __syncthreads();

    // ---- r/z tiles: fused x-proj + h-proj, A-fragment shared by 4 tiles ----
    {
      v8f a0 = {}, a1 = {}, a2 = {}, a3 = {};
#pragma unroll 1
      for (int kt = 0; kt < KT_X; ++kt) {
        const v16bf a = load_a_frag(xA, KPAD, kt, lane);
        const __bf16* b = bx_rz + (size_t)kt * 512;
        a0 = wmma_bf16(a, *(const v16bf*)(b + 0 * KT_X * 512), a0);
        a1 = wmma_bf16(a, *(const v16bf*)(b + 1 * KT_X * 512), a1);
        a2 = wmma_bf16(a, *(const v16bf*)(b + 2 * KT_X * 512), a2);
        a3 = wmma_bf16(a, *(const v16bf*)(b + 3 * KT_X * 512), a3);
      }
#pragma unroll 1
      for (int kt = 0; kt < KT_H; ++kt) {
        const v16bf a = load_a_frag(hA, HH, kt, lane);
        const __bf16* b = bh_rz + (size_t)kt * 512;
        a0 = wmma_bf16(a, *(const v16bf*)(b + 0 * KT_H * 512), a0);
        a1 = wmma_bf16(a, *(const v16bf*)(b + 1 * KT_H * 512), a1);
        a2 = wmma_bf16(a, *(const v16bf*)(b + 2 * KT_H * 512), a2);
        a3 = wmma_bf16(a, *(const v16bf*)(b + 3 * KT_H * 512), a3);
      }
      store_tile(gsum, G3, nt_rz + 0, lane, a0);
      store_tile(gsum, G3, nt_rz + 1, lane, a1);
      store_tile(gsum, G3, nt_rz + 2, lane, a2);
      store_tile(gsum, G3, nt_rz + 3, lane, a3);
    }

    // ---- n tiles: xn and hn kept separate ----
    {
      v8f x0 = {}, x1 = {};
#pragma unroll 1
      for (int kt = 0; kt < KT_X; ++kt) {
        const v16bf a = load_a_frag(xA, KPAD, kt, lane);
        const __bf16* b = bx_n + (size_t)kt * 512;
        x0 = wmma_bf16(a, *(const v16bf*)(b + 0 * KT_X * 512), x0);
        x1 = wmma_bf16(a, *(const v16bf*)(b + 1 * KT_X * 512), x1);
      }
      store_tile(gsum, G3, nt_n + 0, lane, x0);
      store_tile(gsum, G3, nt_n + 1, lane, x1);

      v8f h0 = {}, h1 = {};
#pragma unroll 1
      for (int kt = 0; kt < KT_H; ++kt) {
        const v16bf a = load_a_frag(hA, HH, kt, lane);
        const __bf16* b = bh_n + (size_t)kt * 512;
        h0 = wmma_bf16(a, *(const v16bf*)(b + 0 * KT_H * 512), h0);
        h1 = wmma_bf16(a, *(const v16bf*)(b + 1 * KT_H * 512), h1);
      }
      store_tile(hnb, HH, nt_n - 32 + 0, lane, h0);
      store_tile(hnb, HH, nt_n - 32 + 1, lane, h1);
    }
    __syncthreads();

    // ---- elementwise gates: thread owns column c, loops over 16 rows ----
#pragma unroll
    for (int m = 0; m < 16; ++m) {
      float r  = fast_sigmoid(gsum[m * G3 + c] + brb);
      float z  = fast_sigmoid(gsum[m * G3 + HH + c] + bzb);
      float nn = fast_tanh(gsum[m * G3 + 2 * HH + c] + bxn +
                           r * (hnb[m * HH + c] + bhn));
      float h  = nn + z * (hF[m * HH + c] - nn);   // (1-z)*n + z*h
      hF[m * HH + c] = h;
      hA[m * HH + c] = (__bf16)h;
      if (WRITE_SEQ)
        out_seq[((size_t)(s0 + m) * STEPS + tt) * 512 + dir * HH + c] = h;
    }
    __syncthreads();
  }

  if (!WRITE_SEQ) {
#pragma unroll
    for (int m = 0; m < 16; ++m)
      out_h[(size_t)(s0 + m) * 512 + dir * HH + c] = hF[m * HH + c];
  }
}

// ---------------------------------------------------------------------------
extern "C" void kernel_launch(void* const* d_in, const int* in_sizes, int n_in,
                              void* d_out, int out_size, void* d_ws, size_t ws_size,
                              hipStream_t stream) {
  const int*   tokens = (const int*)  d_in[0];
  const float* embed  = (const float*)d_in[1];
  const float* w_ih_f1 = (const float*)d_in[2];
  const float* w_hh_f1 = (const float*)d_in[3];
  const float* b_ih_f1 = (const float*)d_in[4];
  const float* b_hh_f1 = (const float*)d_in[5];
  const float* w_ih_b1 = (const float*)d_in[6];
  const float* w_hh_b1 = (const float*)d_in[7];
  const float* b_ih_b1 = (const float*)d_in[8];
  const float* b_hh_b1 = (const float*)d_in[9];
  const float* w_ih_f2 = (const float*)d_in[10];
  const float* w_hh_f2 = (const float*)d_in[11];
  const float* b_ih_f2 = (const float*)d_in[12];
  const float* b_hh_f2 = (const float*)d_in[13];
  const float* w_ih_b2 = (const float*)d_in[14];
  const float* w_hh_b2 = (const float*)d_in[15];
  const float* b_ih_b2 = (const float*)d_in[16];
  const float* b_hh_b2 = (const float*)d_in[17];

  __bf16* ws = (__bf16*)d_ws;

  auto pack = [&](const float* W, int off, int Ksrc, int KT) {
    int total = NT_TOT * KT * 512;
    pack_weight<<<(total + 255) / 256, 256, 0, stream>>>(W, ws + off, Ksrc, KT);
  };
  pack(w_ih_f1, OFF_IH_F1, EE, 10);
  pack(w_hh_f1, OFF_HH_F1, HH, 8);
  pack(w_ih_b1, OFF_IH_B1, EE, 10);
  pack(w_hh_b1, OFF_HH_B1, HH, 8);
  pack(w_ih_f2, OFF_IH_F2, 512, 16);
  pack(w_hh_f2, OFF_HH_F2, HH, 8);
  pack(w_ih_b2, OFF_IH_B2, 512, 16);
  pack(w_hh_b2, OFF_HH_B2, HH, 8);

  float* h_ns = (float*)d_out;                        // [1024, 512]
  float* out2 = (float*)d_out + (size_t)BATCH1 * 512; // [32, 32, 512]

  // level 1: word biGRU over 1024 sequences (64 tiles x 2 dirs)
  gru_level<EE, 320, LL, true, false><<<dim3(BATCH1 / 16, 2), 256, 0, stream>>>(
      tokens, embed,
      ws + OFF_IH_F1, ws + OFF_IH_B1, ws + OFF_HH_F1, ws + OFF_HH_B1,
      b_ih_f1, b_ih_b1, b_hh_f1, b_hh_b1,
      h_ns, nullptr);

  // level 2: dialog biGRU over 32 dialogs (2 tiles x 2 dirs)
  gru_level<512, 512, TT_, false, true><<<dim3(DD / 16, 2), 256, 0, stream>>>(
      nullptr, h_ns,
      ws + OFF_IH_F2, ws + OFF_IH_B2, ws + OFF_HH_F2, ws + OFF_HH_B2,
      b_ih_f2, b_ih_b2, b_hh_f2, b_hh_b2,
      nullptr, out2);
}
